// ABCNN_21148418965760
// MI455X (gfx1250) — matrix operand order repaired and checked by reading, not compiled, not run on hardware
//
#include <hip/hip_runtime.h>
#include <hip/hip_bf16.h>

// ---------------------------------------------------------------------------
// Sizes from the reference
// ---------------------------------------------------------------------------
#define BATCH 4096
#define INPUT_DIM 512
#define SEQ 256            // L = INPUT_DIM/2
#define CH 12
#define HID 2048
#define OUTD 128
#define K1 (INPUT_DIM * 12) // 6144, fc1 reduction dim

typedef __attribute__((ext_vector_type(16))) __bf16 v16bf;
typedef __attribute__((ext_vector_type(8)))  float  v8f;

// gfx1250 async global->LDS path (ASYNCcnt), guarded so we fall back to
// global_load + ds_store staging if the toolchain lacks the builtins.
// Probed prototype: (int __vector(4) AS1*, int __vector(4) AS3*, Ii, Ii)
#if defined(__gfx1250__) && \
    __has_builtin(__builtin_amdgcn_global_load_async_to_lds_b128) && \
    __has_builtin(__builtin_amdgcn_s_wait_asynccnt)
#define USE_ASYNC_LDS 1
typedef int async_i4 __attribute__((vector_size(16)));
typedef __attribute__((address_space(1))) async_i4 gbl_i4;
typedef __attribute__((address_space(3))) async_i4 lds_i4;
#else
#define USE_ASYNC_LDS 0
#endif

__device__ __forceinline__ unsigned short f2bf(float f) {
    unsigned int u = __builtin_bit_cast(unsigned int, f);
    u += 0x7FFFu + ((u >> 16) & 1u);   // round-to-nearest-even
    return (unsigned short)(u >> 16);
}

union FragBF {
    uint4 u[2];
    v16bf v;
};

// ---------------------------------------------------------------------------
// Kernel 1: fused conv pipeline. One block per batch row, 256 threads = one
// per sequence position. All intermediates live in LDS.
// Emits bf16 feature rows [BATCH x 6144]: concat(o.reshape, t.reshape).
// ---------------------------------------------------------------------------
__global__ __launch_bounds__(256) void conv_feat_kernel(
    const float* __restrict__ x,
    const float* __restrict__ c1w, const float* __restrict__ c1b,
    const float* __restrict__ c2w, const float* __restrict__ c2b,
    const float* __restrict__ c3w, const float* __restrict__ c3b,
    const float* __restrict__ c4w, const float* __restrict__ c4b,
    unsigned short* __restrict__ Xf)
{
    __shared__ float in_o[SEQ + 2], in_t[SEQ + 2];
    __shared__ float bufA_o[CH * SEQ], bufA_t[CH * SEQ];
    __shared__ float bufB_o[CH * SEQ], bufB_t[CH * SEQ];
    __shared__ float w1[36], b1s[CH];
    __shared__ float w2[432], w3[432], w4[432];
    __shared__ float b2s[CH], b3s[CH], b4s[CH];
    __shared__ float red[256];

    const int p   = threadIdx.x;      // position 0..255
    const int row = blockIdx.x;       // batch row

    if (p < 36) w1[p] = c1w[p];
    if (p < CH) { b1s[p] = c1b[p]; b2s[p] = c2b[p]; b3s[p] = c3b[p]; b4s[p] = c4b[p]; }
    for (int i = p; i < 432; i += 256) { w2[i] = c2w[i]; w3[i] = c3w[i]; w4[i] = c4w[i]; }

    in_o[p + 1] = x[(size_t)row * INPUT_DIM + p];
    in_t[p + 1] = x[(size_t)row * INPUT_DIM + SEQ + p];
    if (p == 0) { in_o[0] = 0.f; in_t[0] = 0.f; in_o[SEQ + 1] = 0.f; in_t[SEQ + 1] = 0.f; }
    __syncthreads();

    // conv1 (1 -> 12 ch) + tanh
    #pragma unroll
    for (int c = 0; c < CH; ++c) {
        float ao = b1s[c], at = b1s[c];
        #pragma unroll
        for (int j = 0; j < 3; ++j) {
            const float w = w1[c * 3 + j];
            ao += w * in_o[p + j];
            at += w * in_t[p + j];
        }
        bufA_o[c * SEQ + p] = tanhf(ao);
        bufA_t[c * SEQ + p] = tanhf(at);
    }
    __syncthreads();

    // s[p] = 1/(1 + sum_c (o-t)^2) ; S = sum_p s[p]
    float sp = 0.f;
    for (int c = 0; c < CH; ++c) {
        const float d = bufA_o[c * SEQ + p] - bufA_t[c * SEQ + p];
        sp += d * d;
    }
    sp = 1.f / (1.f + sp);
    red[p] = sp;
    __syncthreads();
    for (int off = 128; off > 0; off >>= 1) {
        if (p < off) red[p] += red[p + off];
        __syncthreads();
    }
    const float S = red[0];

    // o *= S ;  t = L * s[p] * o
    for (int c = 0; c < CH; ++c) {
        const float o = bufA_o[c * SEQ + p] * S;
        bufA_o[c * SEQ + p] = o;
        bufA_t[c * SEQ + p] = (float)SEQ * sp * o;
    }
    __syncthreads();

    // first pool: avg_pool3 * 3 == neighbor sum (zero padded), bufA -> bufB
    for (int c = 0; c < CH; ++c) {
        float lo = (p > 0)       ? bufA_o[c * SEQ + p - 1] : 0.f;
        float ro = (p < SEQ - 1) ? bufA_o[c * SEQ + p + 1] : 0.f;
        bufB_o[c * SEQ + p] = lo + bufA_o[c * SEQ + p] + ro;
        float lt = (p > 0)       ? bufA_t[c * SEQ + p - 1] : 0.f;
        float rt = (p < SEQ - 1) ? bufA_t[c * SEQ + p + 1] : 0.f;
        bufB_t[c * SEQ + p] = lt + bufA_t[c * SEQ + p] + rt;
    }
    __syncthreads();

    // three (conv 12->12, tanh, avg_pool3) layers; cur = bufB, alt = bufA
    for (int l = 0; l < 3; ++l) {
        const float* W  = (l == 0) ? w2  : (l == 1) ? w3  : w4;
        const float* Bb = (l == 0) ? b2s : (l == 1) ? b3s : b4s;
        for (int c = 0; c < CH; ++c) {
            float ao = Bb[c], at = Bb[c];
            #pragma unroll
            for (int ci = 0; ci < CH; ++ci) {
                const float* wo = &W[(c * CH + ci) * 3];
                const float* io = &bufB_o[ci * SEQ];
                const float* it = &bufB_t[ci * SEQ];
                if (p > 0)       { ao += wo[0] * io[p - 1]; at += wo[0] * it[p - 1]; }
                ao += wo[1] * io[p];  at += wo[1] * it[p];
                if (p < SEQ - 1) { ao += wo[2] * io[p + 1]; at += wo[2] * it[p + 1]; }
            }
            bufA_o[c * SEQ + p] = tanhf(ao);
            bufA_t[c * SEQ + p] = tanhf(at);
        }
        __syncthreads();
        for (int c = 0; c < CH; ++c) {
            float lo = (p > 0)       ? bufA_o[c * SEQ + p - 1] : 0.f;
            float ro = (p < SEQ - 1) ? bufA_o[c * SEQ + p + 1] : 0.f;
            bufB_o[c * SEQ + p] = (lo + bufA_o[c * SEQ + p] + ro) * (1.f / 3.f);
            float lt = (p > 0)       ? bufA_t[c * SEQ + p - 1] : 0.f;
            float rt = (p < SEQ - 1) ? bufA_t[c * SEQ + p + 1] : 0.f;
            bufB_t[c * SEQ + p] = (lt + bufA_t[c * SEQ + p] + rt) * (1.f / 3.f);
        }
        __syncthreads();
    }

    // features (bf16): [row][c*256 + p] for o, offset 3072 for t
    for (int c = 0; c < CH; ++c) {
        Xf[(size_t)row * K1 + c * SEQ + p]        = f2bf(bufB_o[c * SEQ + p]);
        Xf[(size_t)row * K1 + 3072 + c * SEQ + p] = f2bf(bufB_t[c * SEQ + p]);
    }
}

// ---------------------------------------------------------------------------
// Kernel 2: f32 -> bf16 weight conversion
// ---------------------------------------------------------------------------
__global__ void f32_to_bf16_kernel(const float* __restrict__ in,
                                   unsigned short* __restrict__ out, int n)
{
    int i = blockIdx.x * blockDim.x + threadIdx.x;
    if (i < n) out[i] = f2bf(in[i]);
}

// ---------------------------------------------------------------------------
// Kernel 3/4: bf16 WMMA GEMM  C[M,N] = act(A[M,K] @ W[N,K]^T + bias)
// Block tile 128x128, 8 waves (wave32), each wave owns 64x32 = 4x2 WMMA tiles.
// Double-buffered LDS, K stepped by 32 (v_wmma_f32_16x16x32_bf16).
// Stage copies use GLOBAL_LOAD_ASYNC_TO_LDS_B128 when available.
// ---------------------------------------------------------------------------
template<int K, bool RELU, bool OUT_BF16>
__global__ __launch_bounds__(256) void gemm_bf16(
    const unsigned short* __restrict__ A,   // [M x K] bf16
    const unsigned short* __restrict__ W,   // [N x K] bf16
    const float* __restrict__ bias,         // [N]
    void* __restrict__ Cout, int ldc)
{
    constexpr int LDT = 40;                 // padded LDS row stride (elems): 80B, 16B aligned
    __shared__ __align__(16) unsigned short As[2][128 * LDT];
    __shared__ __align__(16) unsigned short Bs[2][128 * LDT];

    const int tid  = threadIdx.x;
    const int lane = tid & 31;
    const int wave = tid >> 5;
    const int wm = wave & 1;                // 0..1 -> 64-row slab
    const int wn = wave >> 1;               // 0..3 -> 32-col slab
    const int bm = blockIdx.x;
    const int bn = blockIdx.y;

    // cooperative load mapping: each thread moves two 16B chunks per tile
    const int lrow = tid >> 2;              // 0..63
    const int lk   = (tid & 3) * 8;         // 0,8,16,24 (elems)
    const unsigned short* Ag = A + (size_t)(bm * 128 + lrow) * K + lk;
    const unsigned short* Wg = W + (size_t)(bn * 128 + lrow) * K + lk;
    const int lA0 = lrow * LDT + lk;
    const int lA1 = (lrow + 64) * LDT + lk;

    v8f acc[4][2];
    #pragma unroll
    for (int i = 0; i < 4; ++i)
        #pragma unroll
        for (int j = 0; j < 2; ++j)
            acc[i][j] = (v8f){0.f, 0.f, 0.f, 0.f, 0.f, 0.f, 0.f, 0.f};

    constexpr int NK = K / 32;

#if USE_ASYNC_LDS
    // ---- async global->LDS staging (no VGPR staging, ASYNCcnt tracked) ----
    {
        __builtin_amdgcn_global_load_async_to_lds_b128(
            (gbl_i4*)Ag, (lds_i4*)&As[0][lA0], 0, 0);
        __builtin_amdgcn_global_load_async_to_lds_b128(
            (gbl_i4*)(Ag + (size_t)64 * K), (lds_i4*)&As[0][lA1], 0, 0);
        __builtin_amdgcn_global_load_async_to_lds_b128(
            (gbl_i4*)Wg, (lds_i4*)&Bs[0][lA0], 0, 0);
        __builtin_amdgcn_global_load_async_to_lds_b128(
            (gbl_i4*)(Wg + (size_t)64 * K), (lds_i4*)&Bs[0][lA1], 0, 0);
        __builtin_amdgcn_s_wait_asynccnt(0);
    }
    __syncthreads();
#else
    // ---- fallback: global_load -> VGPR -> ds_store ----
    {
        uint4 ra0 = *(const uint4*)(Ag);
        uint4 ra1 = *(const uint4*)(Ag + (size_t)64 * K);
        uint4 rb0 = *(const uint4*)(Wg);
        uint4 rb1 = *(const uint4*)(Wg + (size_t)64 * K);
        *(uint4*)&As[0][lA0] = ra0;
        *(uint4*)&As[0][lA1] = ra1;
        *(uint4*)&Bs[0][lA0] = rb0;
        *(uint4*)&Bs[0][lA1] = rb1;
    }
    __syncthreads();
#endif

    int cur = 0;
    const int mbase = wm * 64 + (lane & 15);
    const int kbA   = (lane >> 4) * 8;      // A: lanes 0-15 hold K 0-7 & 16-23; 16-31 hold 8-15 & 24-31
    const int nbase = wn * 32 + (lane & 15);
    const int kbB   = (lane >> 4) * 8;      // B: lanes 0-15 hold K 0-7 & 16-23; 16-31 hold 8-15 & 24-31

    for (int kk = 0; kk < NK; ++kk) {
        const int nxt = cur ^ 1;
#if USE_ASYNC_LDS
        if (kk + 1 < NK) {
            const unsigned short* Agk = Ag + (size_t)(kk + 1) * 32;
            const unsigned short* Wgk = Wg + (size_t)(kk + 1) * 32;
            __builtin_amdgcn_global_load_async_to_lds_b128(
                (gbl_i4*)Agk, (lds_i4*)&As[nxt][lA0], 0, 0);
            __builtin_amdgcn_global_load_async_to_lds_b128(
                (gbl_i4*)(Agk + (size_t)64 * K), (lds_i4*)&As[nxt][lA1], 0, 0);
            __builtin_amdgcn_global_load_async_to_lds_b128(
                (gbl_i4*)Wgk, (lds_i4*)&Bs[nxt][lA0], 0, 0);
            __builtin_amdgcn_global_load_async_to_lds_b128(
                (gbl_i4*)(Wgk + (size_t)64 * K), (lds_i4*)&Bs[nxt][lA1], 0, 0);
        }
#else
        uint4 ra0, ra1, rb0, rb1;
        if (kk + 1 < NK) {
            const unsigned short* Agk = Ag + (size_t)(kk + 1) * 32;
            const unsigned short* Wgk = Wg + (size_t)(kk + 1) * 32;
            ra0 = *(const uint4*)(Agk);
            ra1 = *(const uint4*)(Agk + (size_t)64 * K);
            rb0 = *(const uint4*)(Wgk);
            rb1 = *(const uint4*)(Wgk + (size_t)64 * K);
        }
#endif

        FragBF af[4], bf[2];
        #pragma unroll
        for (int mt = 0; mt < 4; ++mt) {
            const unsigned short* sp = &As[cur][(mbase + mt * 16) * LDT + kbA];
            af[mt].u[0] = *(const uint4*)(sp);
            af[mt].u[1] = *(const uint4*)(sp + 16);
        }
        #pragma unroll
        for (int nt = 0; nt < 2; ++nt) {
            const unsigned short* sp = &Bs[cur][(nbase + nt * 16) * LDT + kbB];
            bf[nt].u[0] = *(const uint4*)(sp);
            bf[nt].u[1] = *(const uint4*)(sp + 16);
        }

        #pragma unroll
        for (int mt = 0; mt < 4; ++mt)
            #pragma unroll
            for (int nt = 0; nt < 2; ++nt)
                acc[mt][nt] = __builtin_amdgcn_wmma_f32_16x16x32_bf16(
                    false, af[mt].v, false, bf[nt].v,
                    (short)0, acc[mt][nt], false, false);

#if USE_ASYNC_LDS
        if (kk + 1 < NK) __builtin_amdgcn_s_wait_asynccnt(0);
#else
        if (kk + 1 < NK) {
            *(uint4*)&As[nxt][lA0] = ra0;
            *(uint4*)&As[nxt][lA1] = ra1;
            *(uint4*)&Bs[nxt][lA0] = rb0;
            *(uint4*)&Bs[nxt][lA1] = rb1;
        }
#endif
        __syncthreads();
        cur = nxt;
    }

    // epilogue: D layout -> lane n = lane&15, VGPR r -> M = r + 8*(lane>=16)
    #pragma unroll
    for (int nt = 0; nt < 2; ++nt) {
        const int n = bn * 128 + wn * 32 + nt * 16 + (lane & 15);
        const float bv = bias[n];
        #pragma unroll
        for (int mt = 0; mt < 4; ++mt) {
            #pragma unroll
            for (int r = 0; r < 8; ++r) {
                const int m = bm * 128 + wm * 64 + mt * 16 + r + 8 * (lane >> 4);
                float v = acc[mt][nt][r] + bv;
                if (RELU) v = v > 0.f ? v : 0.f;
                if (OUT_BF16)
                    ((unsigned short*)Cout)[(size_t)m * ldc + n] = f2bf(v);
                else
                    ((float*)Cout)[(size_t)m * ldc + n] = v;
            }
        }
    }
}

// ---------------------------------------------------------------------------
// Host-side launch
// Workspace layout (bf16 elems):
//   Xf [4096 x 6144]  | W1 [2048 x 6144] | W2 [128 x 2048] | Y1 [4096 x 2048]
//   = ~93 MB total
// ---------------------------------------------------------------------------
extern "C" void kernel_launch(void* const* d_in, const int* in_sizes, int n_in,
                              void* d_out, int out_size, void* d_ws, size_t ws_size,
                              hipStream_t stream) {
    const float* x    = (const float*)d_in[0];
    const float* c1w  = (const float*)d_in[1];
    const float* c1b  = (const float*)d_in[2];
    const float* c2w  = (const float*)d_in[3];
    const float* c2b  = (const float*)d_in[4];
    const float* c3w  = (const float*)d_in[5];
    const float* c3b  = (const float*)d_in[6];
    const float* c4w  = (const float*)d_in[7];
    const float* c4b  = (const float*)d_in[8];
    const float* fc1w = (const float*)d_in[9];
    const float* fc1b = (const float*)d_in[10];
    const float* fc2w = (const float*)d_in[11];
    const float* fc2b = (const float*)d_in[12];

    unsigned short* Xf = (unsigned short*)d_ws;
    unsigned short* W1 = Xf + (size_t)BATCH * K1;
    unsigned short* W2 = W1 + (size_t)HID * K1;
    unsigned short* Y1 = W2 + (size_t)OUTD * HID;

    // 1) fused conv feature extraction -> bf16 features
    conv_feat_kernel<<<BATCH, 256, 0, stream>>>(
        x, c1w, c1b, c2w, c2b, c3w, c3b, c4w, c4b, Xf);

    // 2) weight conversions
    {
        int n1 = HID * K1;
        f32_to_bf16_kernel<<<(n1 + 255) / 256, 256, 0, stream>>>(fc1w, W1, n1);
        int n2 = OUTD * HID;
        f32_to_bf16_kernel<<<(n2 + 255) / 256, 256, 0, stream>>>(fc2w, W2, n2);
    }

    // 3) fc1: Y1 = relu(Xf @ W1^T + b1), bf16 out
    {
        dim3 grid(BATCH / 128, HID / 128);   // 32 x 16
        gemm_bf16<K1, true, true><<<grid, 256, 0, stream>>>(Xf, W1, fc1b, Y1, HID);
    }

    // 4) fc2: out = Y1 @ W2^T + b2, f32 out
    {
        dim3 grid(BATCH / 128, OUTD / 128);  // 32 x 1
        gemm_bf16<HID, false, false><<<grid, 256, 0, stream>>>(Y1, W2, fc2b, d_out, OUTD);
    }
}
